// RecurrentAttention_73916387164397
// MI455X (gfx1250) — matrix operand order repaired and checked by reading, not compiled, not run on hardware
//
#include <hip/hip_runtime.h>
#include <hip/hip_bf16.h>
#include <stdint.h>

// Problem: T=512,B=32,IN=512,H=256,OUT=256. Softmax over t cancels the h term of
// the alignment logits => ctx is step-invariant (see analysis). Recurrence is a
// serial chain of tiny GEMMs: keep weights LDS-resident across all 512 steps.
#define TT   512
#define BB   32
#define INN  512
#define HHN  256
#define ONN  256

#define NWG    16        // cluster-sized grid: one WG per WGP, 2 waves each
#define WGBLK  92160     // per-WG staged block (bytes), contiguous, TDM-loaded
#define WY_OFF    0      // gates-y  B-frags: 4g*8kb*32lane*16 bf16 = 32768 B
#define WH_OFF    32768  // gates-h  B-frags: 32768 B
#define AY_OFF    65536  // attn-y   B-frags: 8kb*32*16 bf16 = 8192 B
#define AH_OFF    73728  // attn-h   B-frags: 8192 B
#define GCTX_OFF  81920  // gate bias frags : 2mt*4g*32lane*8 f32 = 8192 B
#define YB_OFF    90112  // y_base  frags   : 2mt*32lane*8 f32 = 2048 B

// workspace layout (bytes); total ~1.6 MB (assumed <= ws_size)
#define CNT_OFF  0
#define LOG_OFF  256
#define CTX_OFF  65792
#define HX_OFF   131328
#define YX_OFF   147712
#define STG_OFF  164096

typedef __bf16 v16bf __attribute__((ext_vector_type(16)));
typedef __bf16 v8bf  __attribute__((ext_vector_type(8)));
typedef float  v8f   __attribute__((ext_vector_type(8)));
typedef unsigned int v4u  __attribute__((ext_vector_type(4)));
typedef int          v8i_ __attribute__((ext_vector_type(8)));
typedef int          v4i_ __attribute__((ext_vector_type(4)));

__device__ __forceinline__ v8f wmma_bf16(v16bf a, v16bf b, v8f c) {
  return __builtin_amdgcn_wmma_f32_16x16x32_bf16(false, a, false, b, (short)0, c,
                                                 false, false);
}
__device__ __forceinline__ float sigm(float x) { return 1.0f / (1.0f + __expf(-x)); }

// A-fragment (16x32 bf16) from a row-major [.,256] bf16 LDS buffer.
// lane L holds row L&15; K runs [h16, h16+16), h16 = (L>>4)*16.
__device__ __forceinline__ v16bf loadA(const __bf16* base, int row0, int kb, int lane) {
  int r = row0 + (lane & 15);
  int k = kb * 32 + ((lane >> 4) << 4);
  const __bf16* p = base + r * 256 + k;
  v8bf lo = *(const v8bf*)p;
  v8bf hi = *(const v8bf*)(p + 8);
  v16bf a;
#pragma unroll
  for (int i = 0; i < 8; ++i) { a[i] = lo[i]; a[i + 8] = hi[i]; }
  return a;
}
// B-fragments are pre-packed: 16 contiguous bf16 per lane per fragment.
__device__ __forceinline__ v16bf loadB(const __bf16* packed, int frag, int lane) {
  return *(const v16bf*)(packed + (((frag * 32) + lane) << 4));
}

// ---------------- phase 0: init state / counter ----------------
__global__ void ra_init_k(const float* __restrict__ y0, const float* __restrict__ h0,
                          unsigned* cnt, __bf16* hx, __bf16* yx) {
  int gid = blockIdx.x * 256 + threadIdx.x;
  if (gid == 0) *cnt = 0u;
  if (gid < BB * HHN) { hx[gid] = (__bf16)h0[gid]; yx[gid] = (__bf16)y0[gid]; }
}

// ---------------- phase A1: logits[t,b] = inputs[t,b,:].Wa[:IN] ----------------
__global__ void ra_logits_k(const float* __restrict__ inputs, const float* __restrict__ Wa,
                            float* __restrict__ logits) {
  int t = blockIdx.x, tid = threadIdx.x;
  int b = tid >> 3, p = tid & 7;
  const float* row = inputs + ((size_t)t * BB + b) * INN;
  float s = 0.f;
  for (int i = p * 64; i < p * 64 + 64; ++i) s += row[i] * Wa[i];
  __shared__ float red[256];
  red[tid] = s; __syncthreads();
  if (p < 4) red[tid] += red[tid + 4];
  __syncthreads();
  if (p < 2) red[tid] += red[tid + 2];
  __syncthreads();
  if (p == 0) logits[t * BB + b] = red[tid] + red[tid + 1];
}

// ---------------- phase A2: softmax over t + ctx[b,:] ----------------
__global__ void ra_ctx_k(const float* __restrict__ inputs, const float* __restrict__ logits,
                         float* __restrict__ ctx) {
  int b = blockIdx.x, tid = threadIdx.x;
  __shared__ float red[256];
  __shared__ float wsh[TT];
  float l0 = logits[tid * BB + b], l1 = logits[(tid + 256) * BB + b];
  red[tid] = fmaxf(l0, l1); __syncthreads();
  for (int s = 128; s > 0; s >>= 1) { if (tid < s) red[tid] = fmaxf(red[tid], red[tid + s]); __syncthreads(); }
  float mx = red[0]; __syncthreads();
  float e0 = __expf(l0 - mx), e1 = __expf(l1 - mx);
  red[tid] = e0 + e1; __syncthreads();
  for (int s = 128; s > 0; s >>= 1) { if (tid < s) red[tid] += red[tid + s]; __syncthreads(); }
  float inv = 1.f / red[0]; __syncthreads();
  wsh[tid] = e0 * inv; wsh[tid + 256] = e1 * inv; __syncthreads();
  float a0 = 0.f, a1 = 0.f;
  for (int t = 0; t < TT; ++t) {
    float w = wsh[t];
    const float* row = inputs + ((size_t)t * BB + b) * INN;
    a0 += w * row[tid]; a1 += w * row[tid + 256];
  }
  ctx[b * INN + tid] = a0; ctx[b * INN + tid + 256] = a1;
}

// ------- phase B1: g_ctx = ctx@W_ih[:, :IN]^T + b_ih + b_hh ; y_base -------
// written directly in WMMA C/D fragment layout into the per-WG staged blocks.
__global__ void ra_gctx_k(const float* __restrict__ ctx, const float* __restrict__ W_ih,
                          const float* __restrict__ b_ih, const float* __restrict__ b_hh,
                          const float* __restrict__ W_att, const float* __restrict__ b_att,
                          char* __restrict__ stg) {
  int gid = blockIdx.x * 256 + threadIdx.x;
  if (gid >= BB * (4 * HHN + ONN)) return;
  int b = gid / 1280, j = gid % 1280;
  int m = b & 15, mt = b >> 4, e = m & 7, lh = (m >> 3) << 4;
  const float* c = ctx + b * INN;
  if (j < 4 * HHN) {
    const float* wr = W_ih + (size_t)j * (INN + ONN);
    float s = b_ih[j] + b_hh[j];
    for (int i = 0; i < INN; ++i) s += c[i] * wr[i];
    int g = j >> 8, col = j & 255, wg = col >> 4, n = col & 15;
    float* dst = (float*)(stg + (size_t)wg * WGBLK + GCTX_OFF);
    dst[(((mt * 4 + g) * 32) + n + lh) * 8 + e] = s;
  } else {
    int n256 = j - 4 * HHN;
    float s = b_att[n256];
    for (int i = 0; i < INN; ++i) s += c[i] * W_att[(size_t)(ONN + HHN + i) * ONN + n256];
    int wg = n256 >> 4, n = n256 & 15;
    float* dst = (float*)(stg + (size_t)wg * WGBLK + YB_OFF);
    dst[((mt * 32) + n + lh) * 8 + e] = s;
  }
}

// ------- phase B2: pack W_y, W_hh, A_y, A_h into bf16 B-fragment layout -------
__global__ void ra_wpack_k(const float* __restrict__ W_ih, const float* __restrict__ W_hh,
                           const float* __restrict__ W_att, char* __restrict__ stg) {
  int gid = blockIdx.x * 256 + threadIdx.x;        // 16 * 40960 elements
  if (gid >= NWG * 40960) return;
  int wg = gid / 40960, r = gid % 40960;
  __bf16* blk = (__bf16*)(stg + (size_t)wg * WGBLK);
  if (r < 16384) {                                   // W_y = W_ih[:, IN:IN+OUT]
    int idx = r & 15, lane = (r >> 4) & 31, kb = (r >> 9) & 7, g = r >> 12;
    int jj = lane & 15, kk = ((lane >> 4) << 4) + idx;
    float v = W_ih[(size_t)(g * 256 + wg * 16 + jj) * (INN + ONN) + INN + kb * 32 + kk];
    blk[r] = (__bf16)v;
  } else if (r < 32768) {                            // W_hh
    int r2 = r - 16384;
    int idx = r2 & 15, lane = (r2 >> 4) & 31, kb = (r2 >> 9) & 7, g = r2 >> 12;
    int jj = lane & 15, kk = ((lane >> 4) << 4) + idx;
    float v = W_hh[(size_t)(g * 256 + wg * 16 + jj) * HHN + kb * 32 + kk];
    ((__bf16*)((char*)blk + WH_OFF))[r2] = (__bf16)v;
  } else if (r < 36864) {                            // A_y = W_att[0:256, cols]
    int r2 = r - 32768;
    int idx = r2 & 15, lane = (r2 >> 4) & 31, kb = r2 >> 9;
    int n = wg * 16 + (lane & 15), k = kb * 32 + ((lane >> 4) << 4) + idx;
    ((__bf16*)((char*)blk + AY_OFF))[r2] = (__bf16)W_att[(size_t)k * ONN + n];
  } else {                                           // A_h = W_att[256:512, cols]
    int r2 = r - 36864;
    int idx = r2 & 15, lane = (r2 >> 4) & 31, kb = r2 >> 9;
    int n = wg * 16 + (lane & 15), k = kb * 32 + ((lane >> 4) << 4) + idx;
    ((__bf16*)((char*)blk + AH_OFF))[r2] = (__bf16)W_att[(size_t)(ONN + k) * ONN + n];
  }
}

// device-wide step barrier: cluster barrier (fast path / NOP if unclustered)
// plus a monotonic L2 atomic sequence counter (the actual guarantee).
__device__ __forceinline__ void gsync(unsigned* cnt, unsigned& bar) {
  __threadfence();
  __syncthreads();
  if (threadIdx.x < 32) __builtin_amdgcn_s_cluster_barrier();
  if (threadIdx.x == 0) {
    unsigned tgt = (++bar) * NWG;
    __hip_atomic_fetch_add(cnt, 1u, __ATOMIC_ACQ_REL, __HIP_MEMORY_SCOPE_AGENT);
    while (__hip_atomic_load(cnt, __ATOMIC_ACQUIRE, __HIP_MEMORY_SCOPE_AGENT) < tgt)
      __builtin_amdgcn_s_sleep(8);
  }
  __syncthreads();
}

// ---------------- phase C: the serial recurrence, 512 steps ----------------
__global__ void __launch_bounds__(64)
ra_rec_k(const float* __restrict__ c0, const char* __restrict__ stg,
         __bf16* __restrict__ hx, __bf16* __restrict__ yx,
         unsigned* __restrict__ cnt, float* __restrict__ out) {
  extern __shared__ char smem[];                 // 124928 B dynamic LDS
  __bf16* y_lds = (__bf16*)smem;                 // [32][256] bf16
  __bf16* h_lds = (__bf16*)(smem + 16384);       // [32][256] bf16
  const __bf16* Wy   = (const __bf16*)(smem + 32768 + WY_OFF);
  const __bf16* Wh   = (const __bf16*)(smem + 32768 + WH_OFF);
  const __bf16* Ay   = (const __bf16*)(smem + 32768 + AY_OFF);
  const __bf16* Ah   = (const __bf16*)(smem + 32768 + AH_OFF);
  const float*  gctx = (const float*)(smem + 32768 + GCTX_OFF);
  const float*  ybse = (const float*)(smem + 32768 + YB_OFF);

  const int wg = blockIdx.x, tid = threadIdx.x;
  const int lane = tid & 31, mt = tid >> 5;
  const int n = lane & 15, lh = (lane >> 4) * 8;

  // One-shot TDM load of this WG's 90KB staged block into LDS (offset 32768).
  // 6-arg builtin (clang-23 signature): (g0, g1, g2, g3, g4, cpol).
  if (tid < 32) {
    unsigned long long ga = (unsigned long long)(uintptr_t)(stg + (size_t)wg * WGBLK);
    unsigned n8 = WGBLK / 8;                      // 8B units, fits 16-bit tile_dim0
    v4u g0 = { 1u, 32768u, (unsigned)(ga & 0xffffffffu),
               (unsigned)((ga >> 32) & 0x1ffffffu) | 0x80000000u /*type=2*/ };
    v8i_ g1 = { (int)(3u << 16) /*data_size=8B*/,
                (int)((n8 & 0xffffu) << 16) /*tensor_dim0 lo*/,
                (int)((n8 >> 16) | (1u << 16)) /*dim0 hi | tensor_dim1=1*/,
                (int)((n8 & 0xffffu) << 16) /*tile_dim0*/,
                1 /*tile_dim1*/, (int)n8 /*dim0 stride*/, 0, 0 };
    v4i_ gz4 = { 0, 0, 0, 0 };
    v8i_ gz8 = { 0, 0, 0, 0, 0, 0, 0, 0 };
    __builtin_amdgcn_tensor_load_to_lds(g0, g1, gz4, gz4, gz8, 0);
    __builtin_amdgcn_s_wait_tensorcnt(0);
  }
  // initial y/h state into LDS
  {
    const uint4* sy = (const uint4*)yx; const uint4* sh = (const uint4*)hx;
    uint4* dy = (uint4*)y_lds; uint4* dh = (uint4*)h_lds;
#pragma unroll
    for (int k = 0; k < 16; ++k) { dy[tid + k * 64] = sy[tid + k * 64]; dh[tid + k * 64] = sh[tid + k * 64]; }
  }
  __syncthreads();

  // cell-state fragment [16 batch x 16 h-cols], C/D layout
  v8f cfr;
#pragma unroll
  for (int e = 0; e < 8; ++e) cfr[e] = c0[(mt * 16 + lh + e) * HHN + wg * 16 + n];

  unsigned bar = 0;
  for (int t = 0; t < TT; ++t) {
    // gates = g_ctx + y@Wy^T + h@Whh^T   (acc[g] is the [16x16] (mt, gate g) tile)
    v8f acc[4];
#pragma unroll
    for (int g = 0; g < 4; ++g)
      acc[g] = *(const v8f*)(gctx + ((((mt * 4 + g) * 32) + lane) << 3));
#pragma unroll
    for (int kb = 0; kb < 8; ++kb) {
      v16bf aY = loadA(y_lds, mt * 16, kb, lane);
#pragma unroll
      for (int g = 0; g < 4; ++g) acc[g] = wmma_bf16(aY, loadB(Wy, g * 8 + kb, lane), acc[g]);
      v16bf aH = loadA(h_lds, mt * 16, kb, lane);
#pragma unroll
      for (int g = 0; g < 4; ++g) acc[g] = wmma_bf16(aH, loadB(Wh, g * 8 + kb, lane), acc[g]);
    }
    // attention, old-y part (before y_lds is refreshed)
    v8f accY = *(const v8f*)(ybse + (((mt * 32) + lane) << 3));
#pragma unroll
    for (int kb = 0; kb < 8; ++kb)
      accY = wmma_bf16(loadA(y_lds, mt * 16, kb, lane), loadB(Ay, kb, lane), accY);

    // LSTM pointwise on fragments
    v8f hn;
#pragma unroll
    for (int e = 0; e < 8; ++e) {
      float ig = sigm(acc[0][e]), fg = sigm(acc[1][e]);
      float gg = tanhf(acc[2][e]), og = sigm(acc[3][e]);
      float cv = fg * cfr[e] + ig * gg;
      cfr[e] = cv;
      hn[e] = og * tanhf(cv);
    }
    // publish our h' slice, sync, pull full h'
#pragma unroll
    for (int e = 0; e < 8; ++e)
      hx[(mt * 16 + lh + e) * HHN + wg * 16 + n] = (__bf16)hn[e];
    gsync(cnt, bar);
    {
      const uint4* sh = (const uint4*)hx; uint4* dh = (uint4*)h_lds;
#pragma unroll
      for (int k = 0; k < 16; ++k) dh[tid + k * 64] = sh[tid + k * 64];
    }
    __syncthreads();
    // attention, new-h part
#pragma unroll
    for (int kb = 0; kb < 8; ++kb)
      accY = wmma_bf16(loadA(h_lds, mt * 16, kb, lane), loadB(Ah, kb, lane), accY);

    // emit y' (fp32 to output, bf16 to exchange buffer), sync, refresh y_lds
    float* orow = out + (size_t)t * (BB * ONN);
#pragma unroll
    for (int e = 0; e < 8; ++e) {
      int b = mt * 16 + lh + e;
      float v = accY[e];
      orow[b * ONN + wg * 16 + n] = v;
      yx[b * ONN + wg * 16 + n] = (__bf16)v;
    }
    gsync(cnt, bar);
    {
      const uint4* sy = (const uint4*)yx; uint4* dy = (uint4*)y_lds;
#pragma unroll
      for (int k = 0; k < 16; ++k) dy[tid + k * 64] = sy[tid + k * 64];
    }
    __syncthreads();
  }
}

extern "C" void kernel_launch(void* const* d_in, const int* in_sizes, int n_in,
                              void* d_out, int out_size, void* d_ws, size_t ws_size,
                              hipStream_t stream) {
  const float* inputs = (const float*)d_in[0];
  const float* y0     = (const float*)d_in[1];
  const float* h0     = (const float*)d_in[2];
  const float* c0     = (const float*)d_in[3];
  const float* W_a    = (const float*)d_in[4];
  // d_in[5] (b_a) cancels inside the time-softmax -> unused
  const float* W_ih   = (const float*)d_in[6];
  const float* W_hh   = (const float*)d_in[7];
  const float* b_ih   = (const float*)d_in[8];
  const float* b_hh   = (const float*)d_in[9];
  const float* W_att  = (const float*)d_in[10];
  const float* b_att  = (const float*)d_in[11];

  char* ws = (char*)d_ws;                 // needs ~1.64 MB
  unsigned* cnt = (unsigned*)(ws + CNT_OFF);
  float* logits = (float*)(ws + LOG_OFF);
  float* ctx    = (float*)(ws + CTX_OFF);
  __bf16* hx    = (__bf16*)(ws + HX_OFF);
  __bf16* yx    = (__bf16*)(ws + YX_OFF);
  char* stg     = ws + STG_OFF;

  ra_init_k<<<32, 256, 0, stream>>>(y0, h0, cnt, hx, yx);
  ra_logits_k<<<TT, 256, 0, stream>>>(inputs, W_a, logits);
  ra_ctx_k<<<BB, 256, 0, stream>>>(inputs, logits, ctx);
  ra_gctx_k<<<160, 256, 0, stream>>>(ctx, W_ih, b_ih, b_hh, W_att, b_att, stg);
  ra_wpack_k<<<2560, 256, 0, stream>>>(W_ih, W_hh, W_att, stg);
  ra_rec_k<<<NWG, 64, 124928, stream>>>(c0, stg, hx, yx, cnt, (float*)d_out);
}
